// PatchSVAEv2_30090540875902
// MI455X (gfx1250) — compile-verified, matrix operand order checked by reading, not run on hardware
//
#include <hip/hip_runtime.h>
#include <hip/hip_bf16.h>
#include <math.h>

typedef _Float16 half_t;
typedef __attribute__((ext_vector_type(16))) _Float16 v16h;
typedef __attribute__((ext_vector_type(8)))  _Float16 v8h;
typedef __attribute__((ext_vector_type(8)))  float    v8f;

#define NPATCH   8192      // B_*gh*gw = 2*64*64
#define HID      384
#define VMODE    48
#define DDIM     4
#define MODEDIM  57
#define FUSEK    1538      // 4*384+2
#define FUSELD   1568      // ceil32(1538): padded, zero-filled row stride
#define NTOK     4096

__device__ __forceinline__ float gelu_f(float x) {
  return 0.5f * x * (1.0f + erff(x * 0.70710678118654752f));
}

// ---------------------------------------------------------------------------
// f16 WMMA GEMM:  C[M,N] = act( A[M,Kpad] * Wt[Npad,Kpad]^T + bias + R )
//  - A zero-padded to Kpad (mult of 32); M multiple of 64
//  - Wt pre-transposed [Npad][Kpad], Npad = ceil64(N), zero-padded
//  - nsteps >= 2 required (all our Kpads are >= 64)
// Block tile 64x64, 8 waves as 4x2, each wave 16x32 (2 WMMA tiles).
// Double-buffered LDS, register prefetch, last step peeled -> branch-free
// steady-state loop. LDS stride 56 halfs: 16B-aligned, conflict-free.
// ---------------------------------------------------------------------------
#define BM 64
#define BN 64
#define KT 32
#define LDS_S 56

__global__ void __launch_bounds__(256)
gemm_wmma_f16(const half_t* __restrict__ A, int lda,
              const half_t* __restrict__ Wt, int ldwt,
              const float* __restrict__ bias,
              const half_t* __restrict__ R,
              half_t* __restrict__ Ch, float* __restrict__ Cf,
              int ldc, int N, int Kpad, int act)
{
  __shared__ __align__(16) half_t As[2][BM * LDS_S];   // [row][k]
  __shared__ __align__(16) half_t Bs[2][BN * LDS_S];   // [col][k] (transposed)

  const int tid  = threadIdx.x;
  const int lane = tid & 31;
  const int wave = tid >> 5;
  const int wm   = wave >> 1;
  const int wn   = wave & 1;
  const int hi   = lane >> 4;
  const int ml   = lane & 15;
  const int m0   = blockIdx.y * BM;
  const int n0   = blockIdx.x * BN;

  const int lrow = tid >> 2;          // 0..63 (A row / B col)
  const int lkb  = (tid & 3) * 8;     // 0,8,16,24

  const half_t* Aptr = &A [(size_t)(m0 + lrow) * lda  + lkb];
  const half_t* Bptr = &Wt[(size_t)(n0 + lrow) * ldwt + lkb];

  const int abase  = (wm * 16 + ml) * LDS_S + hi * 8;
  const int b0base = (wn * 32 + ml) * LDS_S + hi * 8;
  const int b1base = b0base + 16 * LDS_S;
  const int sbase  = lrow * LDS_S + lkb;

  // stage tile 0
  {
    v8h va = *(const v8h*)Aptr;
    v8h vb = *(const v8h*)Bptr;
    *(v8h*)&As[0][sbase] = va;
    *(v8h*)&Bs[0][sbase] = vb;
  }

  v8f acc0 = {}, acc1 = {};
  const int nsteps = Kpad >> 5;   // >= 2

  // steady state: unconditional prefetch, branch-free body
  for (int s = 0; s < nsteps - 1; ++s) {
    const int cur = s & 1;
    __syncthreads();

    v8h va = *(const v8h*)(Aptr + (size_t)(s + 1) * KT);
    v8h vb = *(const v8h*)(Bptr + (size_t)(s + 1) * KT);

    v8h a_lo = *(const v8h*)&As[cur][abase];
    v8h a_hi = *(const v8h*)&As[cur][abase + 16];
    v16h af = __builtin_shufflevector(a_lo, a_hi,
              0,1,2,3,4,5,6,7,8,9,10,11,12,13,14,15);
    v8h b0lo = *(const v8h*)&Bs[cur][b0base];
    v8h b0hi = *(const v8h*)&Bs[cur][b0base + 16];
    v16h bf0 = __builtin_shufflevector(b0lo, b0hi,
               0,1,2,3,4,5,6,7,8,9,10,11,12,13,14,15);
    v8h b1lo = *(const v8h*)&Bs[cur][b1base];
    v8h b1hi = *(const v8h*)&Bs[cur][b1base + 16];
    v16h bf1 = __builtin_shufflevector(b1lo, b1hi,
               0,1,2,3,4,5,6,7,8,9,10,11,12,13,14,15);

    acc0 = __builtin_amdgcn_wmma_f32_16x16x32_f16(false, af, false, bf0, (short)0, acc0, false, false);
    acc1 = __builtin_amdgcn_wmma_f32_16x16x32_f16(false, af, false, bf1, (short)0, acc1, false, false);

    *(v8h*)&As[cur ^ 1][sbase] = va;
    *(v8h*)&Bs[cur ^ 1][sbase] = vb;
  }

  // final (peeled) step: compute only
  {
    const int cur = (nsteps - 1) & 1;
    __syncthreads();
    v8h a_lo = *(const v8h*)&As[cur][abase];
    v8h a_hi = *(const v8h*)&As[cur][abase + 16];
    v16h af = __builtin_shufflevector(a_lo, a_hi,
              0,1,2,3,4,5,6,7,8,9,10,11,12,13,14,15);
    v8h b0lo = *(const v8h*)&Bs[cur][b0base];
    v8h b0hi = *(const v8h*)&Bs[cur][b0base + 16];
    v16h bf0 = __builtin_shufflevector(b0lo, b0hi,
               0,1,2,3,4,5,6,7,8,9,10,11,12,13,14,15);
    v8h b1lo = *(const v8h*)&Bs[cur][b1base];
    v8h b1hi = *(const v8h*)&Bs[cur][b1base + 16];
    v16h bf1 = __builtin_shufflevector(b1lo, b1hi,
               0,1,2,3,4,5,6,7,8,9,10,11,12,13,14,15);
    acc0 = __builtin_amdgcn_wmma_f32_16x16x32_f16(false, af, false, bf0, (short)0, acc0, false, false);
    acc1 = __builtin_amdgcn_wmma_f32_16x16x32_f16(false, af, false, bf1, (short)0, acc1, false, false);
  }

  // epilogue
  #pragma unroll
  for (int j = 0; j < 8; ++j) {
    int row  = m0 + wm * 16 + hi * 8 + j;
    int col0 = n0 + wn * 32 + ml;
    int col1 = col0 + 16;
    if (col0 < N) {
      float v = acc0[j];
      if (bias) v += bias[col0];
      if (R)    v += (float)R[(size_t)row * ldc + col0];
      if (act == 1) v = gelu_f(v);
      if (Ch) Ch[(size_t)row * ldc + col0] = (half_t)v;
      if (Cf) Cf[(size_t)row * ldc + col0] = v;
    }
    if (col1 < N) {
      float v = acc1[j];
      if (bias) v += bias[col1];
      if (R)    v += (float)R[(size_t)row * ldc + col1];
      if (act == 1) v = gelu_f(v);
      if (Ch) Ch[(size_t)row * ldc + col1] = (half_t)v;
      if (Cf) Cf[(size_t)row * ldc + col1] = v;
    }
  }
}

// ---------------------------------------------------------------------------
// LayerNorm (wave per row), f16 in/out, f32 params
// ---------------------------------------------------------------------------
__global__ void __launch_bounds__(256)
ln_f16(const half_t* __restrict__ X, half_t* __restrict__ Y,
       const float* __restrict__ g, const float* __restrict__ b,
       int rows, int n, int ld)
{
  int wave = threadIdx.x >> 5, lane = threadIdx.x & 31;
  int row = blockIdx.x * 8 + wave;
  if (row >= rows) return;
  const half_t* x = X + (size_t)row * ld;
  float s = 0.f, s2 = 0.f;
  for (int i = lane; i < n; i += 32) { float v = (float)x[i]; s += v; s2 += v * v; }
  #pragma unroll
  for (int o = 16; o > 0; o >>= 1) { s += __shfl_xor(s, o, 32); s2 += __shfl_xor(s2, o, 32); }
  float mu  = s / (float)n;
  float var = s2 / (float)n - mu * mu;
  float inv = rsqrtf(var + 1e-5f);
  half_t* y = Y + (size_t)row * ld;
  for (int i = lane; i < n; i += 32)
    y[i] = (half_t)(((float)x[i] - mu) * inv * g[i] + b[i]);
}

// ---------------------------------------------------------------------------
// weight convert + transpose + pad:  src[K,N] f32 -> dst[Npad,Kpad] f16
// ---------------------------------------------------------------------------
__global__ void wt_k(const float* __restrict__ src, half_t* __restrict__ dst,
                     int K, int N, int Kpad, int Npad)
{
  int idx = blockIdx.x * blockDim.x + threadIdx.x;
  if (idx >= Npad * Kpad) return;
  int n = idx / Kpad, k = idx % Kpad;
  dst[idx] = (k < K && n < N) ? (half_t)src[(size_t)k * N + n] : (half_t)0.f;
}

// patchify with K-pad to 64 columns
__global__ void patchify_k(const float* __restrict__ img, half_t* __restrict__ X) {
  int idx = blockIdx.x * blockDim.x + threadIdx.x;
  if (idx >= NPATCH * 64) return;
  int p = idx / 64, col = idx % 64;
  if (col >= VMODE) { X[idx] = (half_t)0.f; return; }
  int b = p / NTOK, rem = p % NTOK;
  int gy = rem / 64, gx = rem % 64;
  int c = col / 16, r2 = col % 16;
  int py = r2 / 4, px = r2 % 4;
  X[idx] = (half_t)img[(((size_t)b * 3 + c) * 256 + gy * 4 + py) * 256 + gx * 4 + px];
}

// per-patch: normalize rows of M, Gram, 4x4 Jacobi eigh, derived scalars
__global__ void eig4_k(const float* __restrict__ Mf,
                       float* __restrict__ VmO, float* __restrict__ SvO,
                       float* __restrict__ evO, float* __restrict__ chO,
                       float* __restrict__ rfO, float* __restrict__ S0)
{
  int p = blockIdx.x * blockDim.x + threadIdx.x;
  if (p >= NPATCH) return;
  const float* Mp = Mf + (size_t)p * (VMODE * DDIM);
  float G[4][4] = {};
  for (int v = 0; v < VMODE; ++v) {
    float m[4], nn = 0.f;
    #pragma unroll
    for (int d = 0; d < 4; ++d) { m[d] = Mp[v * 4 + d]; nn += m[d] * m[d]; }
    float inv = 1.f / fmaxf(sqrtf(nn), 1e-12f);
    #pragma unroll
    for (int d = 0; d < 4; ++d) m[d] *= inv;
    #pragma unroll
    for (int d = 0; d < 4; ++d)
      #pragma unroll
      for (int e = 0; e < 4; ++e) G[d][e] += m[d] * m[e];
  }
  float G0[4][4], V[4][4];
  #pragma unroll
  for (int i = 0; i < 4; ++i)
    #pragma unroll
    for (int j = 0; j < 4; ++j) { G0[i][j] = G[i][j]; V[i][j] = (i == j) ? 1.f : 0.f; }

  const int pr[6] = {0, 0, 0, 1, 1, 2};
  const int qr[6] = {1, 2, 3, 2, 3, 3};
  for (int sweep = 0; sweep < 10; ++sweep) {
    for (int t = 0; t < 6; ++t) {
      int pp = pr[t], qq = qr[t];
      float apq = G[pp][qq];
      if (fabsf(apq) < 1e-20f) continue;
      float tau = (G[qq][qq] - G[pp][pp]) / (2.f * apq);
      float tt  = ((tau >= 0.f) ? 1.f : -1.f) / (fabsf(tau) + sqrtf(1.f + tau * tau));
      float c = rsqrtf(1.f + tt * tt), s = tt * c;
      #pragma unroll
      for (int k = 0; k < 4; ++k) {
        float gkp = G[k][pp], gkq = G[k][qq];
        G[k][pp] = c * gkp - s * gkq;
        G[k][qq] = s * gkp + c * gkq;
      }
      #pragma unroll
      for (int k = 0; k < 4; ++k) {
        float gpk = G[pp][k], gqk = G[qq][k];
        G[pp][k] = c * gpk - s * gqk;
        G[qq][k] = s * gpk + c * gqk;
      }
      #pragma unroll
      for (int k = 0; k < 4; ++k) {
        float vkp = V[k][pp], vkq = V[k][qq];
        V[k][pp] = c * vkp - s * vkq;
        V[k][qq] = s * vkp + c * vkq;
      }
    }
  }
  float ev[4]; int ord[4];
  #pragma unroll
  for (int i = 0; i < 4; ++i) { ev[i] = G[i][i]; ord[i] = i; }
  for (int i = 0; i < 3; ++i)
    for (int j = i + 1; j < 4; ++j)
      if (ev[ord[j]] > ev[ord[i]]) { int tmp = ord[i]; ord[i] = ord[j]; ord[j] = tmp; }

  float rsum = 0.f;
  for (int j = 0; j < 4; ++j) {
    int cj = ord[j];
    float e = ev[cj];
    evO[p * 4 + j] = e;
    SvO[p * 4 + j] = sqrtf(fmaxf(e, 1e-24f));
    S0[p * 4 + j]  = SvO[p * 4 + j];
    chO[p * 4 + j] = G0[j][j];
    for (int i = 0; i < 4; ++i) {
      float vij = V[i][cj];
      VmO[p * 16 + i * 4 + j] = vij;
      float t = 0.f;
      #pragma unroll
      for (int k = 0; k < 4; ++k) t += G0[i][k] * V[k][cj];
      t -= vij * e;
      rsum += t * t;
    }
  }
  rfO[p] = sqrtf(rsum);
}

// U = normalize(M) @ Vm / clip(Sv,1e-16), stored f16 [P,48,4]
__global__ void ukern(const float* __restrict__ Mf, const float* __restrict__ Vm,
                      const float* __restrict__ Sv, half_t* __restrict__ Uh)
{
  int idx = blockIdx.x * blockDim.x + threadIdx.x;
  if (idx >= NPATCH * VMODE) return;
  int p = idx / VMODE, v = idx % VMODE;
  float m[4], nn = 0.f;
  #pragma unroll
  for (int d = 0; d < 4; ++d) { m[d] = Mf[(size_t)p * 192 + v * 4 + d]; nn += m[d] * m[d]; }
  float inv = 1.f / fmaxf(sqrtf(nn), 1e-12f);
  #pragma unroll
  for (int d = 0; d < 4; ++d) m[d] *= inv;
  #pragma unroll
  for (int e = 0; e < 4; ++e) {
    float dot = 0.f;
    #pragma unroll
    for (int d = 0; d < 4; ++d) dot += m[d] * Vm[p * 16 + d * 4 + e];
    Uh[((size_t)p * VMODE + v) * 4 + e] = (half_t)(dot / fmaxf(Sv[p * 4 + e], 1e-16f));
  }
}

// cross-attn pieces
__global__ void qkv_k(const float* __restrict__ S, const float* __restrict__ g,
                      const float* __restrict__ bn, const float* __restrict__ Wq,
                      const float* __restrict__ bq, float* __restrict__ QKV)
{
  int t = blockIdx.x * blockDim.x + threadIdx.x;
  if (t >= NPATCH) return;
  float x[4], mu = 0.f;
  #pragma unroll
  for (int d = 0; d < 4; ++d) { x[d] = S[t * 4 + d]; mu += x[d]; }
  mu *= 0.25f;
  float var = 0.f;
  #pragma unroll
  for (int d = 0; d < 4; ++d) { float c = x[d] - mu; var += c * c; }
  var *= 0.25f;
  float inv = rsqrtf(var + 1e-5f);
  float sn[4];
  #pragma unroll
  for (int d = 0; d < 4; ++d) sn[d] = (x[d] - mu) * inv * g[d] + bn[d];
  for (int j = 0; j < 12; ++j) {
    float a = bq[j];
    #pragma unroll
    for (int d = 0; d < 4; ++d) a += sn[d] * Wq[d * 12 + j];
    QKV[(size_t)t * 12 + j] = a;
  }
}

__global__ void attn_k(const float* __restrict__ QKV, float* __restrict__ O)
{
  int idx = blockIdx.x * blockDim.x + threadIdx.x;
  if (idx >= 2 * 2 * NTOK) return;
  int n = idx % NTOK;
  int h = (idx / NTOK) & 1;
  int b = idx / (2 * NTOK);
  const float* base = QKV + (size_t)b * NTOK * 12;
  float q0 = base[(size_t)n * 12 + h * 2 + 0];
  float q1 = base[(size_t)n * 12 + h * 2 + 1];
  const float scale = 0.70710678f;
  float mx = -1e30f, se = 0.f, a0 = 0.f, a1 = 0.f;
  for (int m = 0; m < NTOK; ++m) {
    float k0 = base[(size_t)m * 12 + 4 + h * 2];
    float k1 = base[(size_t)m * 12 + 5 + h * 2];
    float sc = (q0 * k0 + q1 * k1) * scale;
    float nm = fmaxf(mx, sc);
    float corr = __expf(mx - nm);
    float e    = __expf(sc - nm);
    float v0 = base[(size_t)m * 12 + 8 + h * 2];
    float v1 = base[(size_t)m * 12 + 9 + h * 2];
    se = se * corr + e;
    a0 = a0 * corr + e * v0;
    a1 = a1 * corr + e * v1;
    mx = nm;
  }
  float inv = 1.f / se;
  O[((size_t)b * NTOK + n) * 4 + h * 2 + 0] = a0 * inv;
  O[((size_t)b * NTOK + n) * 4 + h * 2 + 1] = a1 * inv;
}

__global__ void gate_k(const float* __restrict__ S, const float* __restrict__ O,
                       const float* __restrict__ Wo, const float* __restrict__ bo,
                       const float* __restrict__ al, float* __restrict__ Sn)
{
  int t = blockIdx.x * blockDim.x + threadIdx.x;
  if (t >= NPATCH) return;
  float o[4];
  #pragma unroll
  for (int d = 0; d < 4; ++d) o[d] = O[t * 4 + d];
  #pragma unroll
  for (int d = 0; d < 4; ++d) {
    float gv = bo[d];
    #pragma unroll
    for (int e = 0; e < 4; ++e) gv += o[e] * Wo[e * 4 + d];
    gv = tanhf(gv);
    float a = 0.2f / (1.f + __expf(-al[d]));
    Sn[t * 4 + d] = S[t * 4 + d] * (1.f + a * gv);
  }
}

// per-(p,d) bundle row of 57 halfs, zero-padded to stride 64
__global__ void bundle_k(const half_t* __restrict__ Uh, const float* __restrict__ S,
                         const float* __restrict__ Vm, const float* __restrict__ ev,
                         const float* __restrict__ Sv, const float* __restrict__ ch,
                         const float* __restrict__ rf, half_t* __restrict__ Bn)
{
  int r = blockIdx.x * blockDim.x + threadIdx.x;
  if (r >= NPATCH * 4) return;
  int p = r >> 2, d = r & 3;
  half_t* out = Bn + (size_t)r * 64;
  for (int v = 0; v < VMODE; ++v) out[v] = Uh[((size_t)p * VMODE + v) * 4 + d];
  out[48] = (half_t)S[p * 4 + d];
  #pragma unroll
  for (int f = 0; f < 4; ++f) out[49 + f] = (half_t)Vm[p * 16 + f * 4 + d];
  out[53] = (half_t)log1pf(fmaxf(ev[p * 4 + d], 0.f));
  out[54] = (half_t)Sv[p * 4 + d];
  out[55] = (half_t)ch[p * 4 + d];
  out[56] = (half_t)(float)d;
  #pragma unroll
  for (int j = 57; j < 64; ++j) out[j] = (half_t)0.f;
}

// fused = [ mhn viewed as (P,1536) | refine | 0-pad ] row stride FUSELD=1568
__global__ void fused_k(const half_t* __restrict__ mhn, const float* __restrict__ rf,
                        half_t* __restrict__ F)
{
  size_t idx = (size_t)blockIdx.x * blockDim.x + threadIdx.x;
  if (idx >= (size_t)NPATCH * FUSELD) return;
  int p = (int)(idx / FUSELD), c = (int)(idx % FUSELD);
  half_t v;
  if (c < 1536)       v = mhn[(size_t)p * 1536 + c];
  else if (c == 1536) v = (half_t)rf[p];
  else                v = (half_t)0.f;
  F[idx] = v;
}

// Op has padded row stride 64 (cols 48..63 junk, never read)
__global__ void unpatch_k(const float* __restrict__ outp, float* __restrict__ img)
{
  int idx = blockIdx.x * blockDim.x + threadIdx.x;
  if (idx >= 2 * 3 * 256 * 256) return;
  int b = idx / (3 * 65536);
  int c = (idx / 65536) % 3;
  int y = (idx / 256) % 256;
  int x = idx % 256;
  int p = b * NTOK + (y >> 2) * 64 + (x >> 2);
  int col = c * 16 + (y & 3) * 4 + (x & 3);
  img[idx] = outp[(size_t)p * 64 + col];
}

__global__ void conv1_k(const float* __restrict__ img, const float* __restrict__ w,
                        const float* __restrict__ bias, float* __restrict__ y)
{
  int idx = blockIdx.x * blockDim.x + threadIdx.x;
  if (idx >= 2 * 8 * 256 * 256) return;
  int b = idx / (8 * 65536);
  int o = (idx / 65536) % 8;
  int yy = (idx / 256) % 256;
  int xx = idx % 256;
  float acc = bias[o];
  for (int i = 0; i < 3; ++i)
    #pragma unroll
    for (int ky = 0; ky < 3; ++ky) {
      int sy = yy + ky - 1;
      if ((unsigned)sy >= 256u) continue;
      #pragma unroll
      for (int kx = 0; kx < 3; ++kx) {
        int sx = xx + kx - 1;
        if ((unsigned)sx >= 256u) continue;
        acc += img[(((size_t)b * 3 + i) * 256 + sy) * 256 + sx] *
               w[((o * 3 + i) * 3 + ky) * 3 + kx];
      }
    }
  y[idx] = gelu_f(acc);
}

__global__ void conv2_k(const float* __restrict__ y1, const float* __restrict__ w,
                        const float* __restrict__ bias, const float* __restrict__ img,
                        float* __restrict__ out)
{
  int idx = blockIdx.x * blockDim.x + threadIdx.x;
  if (idx >= 2 * 3 * 256 * 256) return;
  int b = idx / (3 * 65536);
  int o = (idx / 65536) % 3;
  int yy = (idx / 256) % 256;
  int xx = idx % 256;
  float acc = bias[o];
  for (int i = 0; i < 8; ++i)
    #pragma unroll
    for (int ky = 0; ky < 3; ++ky) {
      int sy = yy + ky - 1;
      if ((unsigned)sy >= 256u) continue;
      #pragma unroll
      for (int kx = 0; kx < 3; ++kx) {
        int sx = xx + kx - 1;
        if ((unsigned)sx >= 256u) continue;
        acc += y1[(((size_t)b * 8 + i) * 256 + sy) * 256 + sx] *
               w[((o * 8 + i) * 3 + ky) * 3 + kx];
      }
    }
  out[idx] = img[idx] + acc;
}

// ---------------------------------------------------------------------------
// host-side orchestration
// ---------------------------------------------------------------------------
static inline dim3 gemm_grid(int M, int N) { return dim3((N + BN - 1) / BN, (M + BM - 1) / BM); }
static inline int ceil64(int n) { return (n + 63) & ~63; }

extern "C" void kernel_launch(void* const* d_in, const int* in_sizes, int n_in,
                              void* d_out, int out_size, void* d_ws, size_t ws_size,
                              hipStream_t stream)
{
  (void)in_sizes; (void)out_size; (void)ws_size;
  if (n_in < 81) return;

  const float* images   = (const float*)d_in[0];
  const float* encin_w  = (const float*)d_in[1];
  const float* encin_b  = (const float*)d_in[2];
  const float* encout_w = (const float*)d_in[27];
  const float* encout_b = (const float*)d_in[28];
  const float* proj_w   = (const float*)d_in[43];
  const float* proj_b   = (const float*)d_in[44];
  const float* pn_g     = (const float*)d_in[45];
  const float* pn_b     = (const float*)d_in[46];
  const float* fuse_w   = (const float*)d_in[47];
  const float* fuse_b   = (const float*)d_in[48];
  const float* fn_g     = (const float*)d_in[49];
  const float* fn_b     = (const float*)d_in[50];
  const float* outw     = (const float*)d_in[75];
  const float* outb     = (const float*)d_in[76];
  const float* c1w = (const float*)d_in[77];
  const float* c1b = (const float*)d_in[78];
  const float* c2w = (const float*)d_in[79];
  const float* c2b = (const float*)d_in[80];

  char* base = (char*)d_ws;
  size_t off = 0;
  auto alloc = [&](size_t bytes) -> void* {
    size_t a = (off + 255) & ~(size_t)255;
    off = a + bytes;
    return (void*)(base + a);
  };

  // transposed, padded f16 weights: Wt[Npad][Kpad]
  half_t* Wenc_in  = (half_t*)alloc((size_t)HID * 64 * 2);             // K 48->64
  half_t* Wel1[4], *Wel2[4], *Wdl1[4], *Wdl2[4];
  for (int i = 0; i < 4; ++i) Wel1[i] = (half_t*)alloc((size_t)HID * HID * 2);
  for (int i = 0; i < 4; ++i) Wel2[i] = (half_t*)alloc((size_t)HID * HID * 2);
  for (int i = 0; i < 4; ++i) Wdl1[i] = (half_t*)alloc((size_t)HID * HID * 2);
  for (int i = 0; i < 4; ++i) Wdl2[i] = (half_t*)alloc((size_t)HID * HID * 2);
  half_t* Wenc_out = (half_t*)alloc((size_t)(VMODE * DDIM) * HID * 2); // N=192 mult64
  half_t* Wproj    = (half_t*)alloc((size_t)HID * 64 * 2);             // K 57->64
  half_t* Wfuse    = (half_t*)alloc((size_t)HID * FUSELD * 2);         // K 1538->1568
  half_t* Wout     = (half_t*)alloc((size_t)64 * HID * 2);             // N 48->64

  half_t* X0  = (half_t*)alloc((size_t)NPATCH * 64 * 2);
  half_t* Ha  = (half_t*)alloc((size_t)NPATCH * HID * 2);
  half_t* Hb  = (half_t*)alloc((size_t)NPATCH * HID * 2);
  half_t* Tb  = (half_t*)alloc((size_t)NPATCH * HID * 2);
  half_t* T1  = (half_t*)alloc((size_t)NPATCH * HID * 2);
  float*  Mf  = (float*)alloc((size_t)NPATCH * VMODE * DDIM * 4);
  float*  Vm  = (float*)alloc((size_t)NPATCH * 16 * 4);
  float*  Sv  = (float*)alloc((size_t)NPATCH * 4 * 4);
  float*  Ev  = (float*)alloc((size_t)NPATCH * 4 * 4);
  float*  Ch  = (float*)alloc((size_t)NPATCH * 4 * 4);
  float*  Rf  = (float*)alloc((size_t)NPATCH * 4);
  float*  Sa  = (float*)alloc((size_t)NPATCH * 4 * 4);
  float*  Sb  = (float*)alloc((size_t)NPATCH * 4 * 4);
  float*  QKV = (float*)alloc((size_t)NPATCH * 12 * 4);
  float*  AO  = (float*)alloc((size_t)NPATCH * 4 * 4);
  half_t* Uh  = (half_t*)alloc((size_t)NPATCH * VMODE * DDIM * 2);
  half_t* Bn  = (half_t*)alloc((size_t)NPATCH * 4 * 64 * 2);
  half_t* Mh  = (half_t*)alloc((size_t)NPATCH * 4 * HID * 2);
  half_t* Mhn = (half_t*)alloc((size_t)NPATCH * 4 * HID * 2);
  half_t* Fu  = (half_t*)alloc((size_t)NPATCH * FUSELD * 2);
  float*  Op  = (float*)alloc((size_t)NPATCH * 64 * 4);                // ldc 64 (N=48 padded)
  float*  Img = (float*)alloc((size_t)2 * 3 * 256 * 256 * 4);
  float*  Y1  = (float*)alloc((size_t)2 * 8 * 256 * 256 * 4);

  auto wt = [&](const float* s, half_t* d, int K, int N, int Kpad) {
    int Npad = ceil64(N);
    int n = Npad * Kpad;
    wt_k<<<(n + 255) / 256, 256, 0, stream>>>(s, d, K, N, Kpad, Npad);
  };

  wt(encin_w, Wenc_in, VMODE, HID, 64);
  for (int i = 0; i < 4; ++i) {
    wt((const float*)d_in[3 + i * 6 + 2], Wel1[i], HID, HID, HID);
    wt((const float*)d_in[3 + i * 6 + 4], Wel2[i], HID, HID, HID);
    wt((const float*)d_in[51 + i * 6 + 2], Wdl1[i], HID, HID, HID);
    wt((const float*)d_in[51 + i * 6 + 4], Wdl2[i], HID, HID, HID);
  }
  wt(encout_w, Wenc_out, HID, VMODE * DDIM, HID);
  wt(proj_w, Wproj, MODEDIM, HID, 64);
  wt(fuse_w, Wfuse, FUSEK, HID, FUSELD);
  wt(outw, Wout, HID, VMODE, HID);

  // ---- encoder ----
  patchify_k<<<(NPATCH * 64 + 255) / 256, 256, 0, stream>>>(images, X0);

  gemm_wmma_f16<<<gemm_grid(NPATCH, HID), 256, 0, stream>>>(
      X0, 64, Wenc_in, 64, encin_b, nullptr, Ha, nullptr, HID, HID, 64, 1);

  half_t* Hc = Ha; half_t* Hn = Hb;
  for (int i = 0; i < 4; ++i) {
    const float* lng = (const float*)d_in[3 + i * 6 + 0];
    const float* lnb = (const float*)d_in[3 + i * 6 + 1];
    const float* b1  = (const float*)d_in[3 + i * 6 + 3];
    const float* b2  = (const float*)d_in[3 + i * 6 + 5];
    ln_f16<<<(NPATCH + 7) / 8, 256, 0, stream>>>(Hc, Tb, lng, lnb, NPATCH, HID, HID);
    gemm_wmma_f16<<<gemm_grid(NPATCH, HID), 256, 0, stream>>>(
        Tb, HID, Wel1[i], HID, b1, nullptr, T1, nullptr, HID, HID, HID, 1);
    gemm_wmma_f16<<<gemm_grid(NPATCH, HID), 256, 0, stream>>>(
        T1, HID, Wel2[i], HID, b2, Hc, Hn, nullptr, HID, HID, HID, 0);
    half_t* t = Hc; Hc = Hn; Hn = t;
  }

  gemm_wmma_f16<<<gemm_grid(NPATCH, VMODE * DDIM), 256, 0, stream>>>(
      Hc, HID, Wenc_out, HID, encout_b, nullptr, nullptr, Mf,
      VMODE * DDIM, VMODE * DDIM, HID, 0);

  // ---- per-patch eigendecomposition + U ----
  eig4_k<<<(NPATCH + 255) / 256, 256, 0, stream>>>(Mf, Vm, Sv, Ev, Ch, Rf, Sa);
  ukern<<<(NPATCH * VMODE + 255) / 256, 256, 0, stream>>>(Mf, Vm, Sv, Uh);

  // ---- cross-attention (2 layers) ----
  float* Scur = Sa; float* Snew = Sb;
  for (int j = 0; j < 2; ++j) {
    int bidx = 29 + j * 7;
    qkv_k<<<(NPATCH + 255) / 256, 256, 0, stream>>>(
        Scur, (const float*)d_in[bidx + 0], (const float*)d_in[bidx + 1],
        (const float*)d_in[bidx + 2], (const float*)d_in[bidx + 3], QKV);
    attn_k<<<(2 * 2 * NTOK + 255) / 256, 256, 0, stream>>>(QKV, AO);
    gate_k<<<(NPATCH + 255) / 256, 256, 0, stream>>>(
        Scur, AO, (const float*)d_in[bidx + 4], (const float*)d_in[bidx + 5],
        (const float*)d_in[bidx + 6], Snew);
    float* t = Scur; Scur = Snew; Snew = t;
  }

  // ---- decoder ----
  bundle_k<<<(NPATCH * 4 + 255) / 256, 256, 0, stream>>>(Uh, Scur, Vm, Ev, Sv, Ch, Rf, Bn);

  gemm_wmma_f16<<<gemm_grid(NPATCH * 4, HID), 256, 0, stream>>>(
      Bn, 64, Wproj, 64, proj_b, nullptr, Mh, nullptr, HID, HID, 64, 1);
  ln_f16<<<(NPATCH * 4 + 7) / 8, 256, 0, stream>>>(Mh, Mhn, pn_g, pn_b, NPATCH * 4, HID, HID);

  {
    size_t tot = (size_t)NPATCH * FUSELD;
    fused_k<<<(unsigned)((tot + 255) / 256), 256, 0, stream>>>(Mhn, Rf, Fu);
  }

  gemm_wmma_f16<<<gemm_grid(NPATCH, HID), 256, 0, stream>>>(
      Fu, FUSELD, Wfuse, FUSELD, fuse_b, nullptr, T1, nullptr, HID, HID, FUSELD, 1);
  ln_f16<<<(NPATCH + 7) / 8, 256, 0, stream>>>(T1, Ha, fn_g, fn_b, NPATCH, HID, HID);

  Hc = Ha; Hn = Hb;
  for (int i = 0; i < 4; ++i) {
    const float* lng = (const float*)d_in[51 + i * 6 + 0];
    const float* lnb = (const float*)d_in[51 + i * 6 + 1];
    const float* b1  = (const float*)d_in[51 + i * 6 + 3];
    const float* b2  = (const float*)d_in[51 + i * 6 + 5];
    ln_f16<<<(NPATCH + 7) / 8, 256, 0, stream>>>(Hc, Tb, lng, lnb, NPATCH, HID, HID);
    gemm_wmma_f16<<<gemm_grid(NPATCH, HID), 256, 0, stream>>>(
        Tb, HID, Wdl1[i], HID, b1, nullptr, T1, nullptr, HID, HID, HID, 1);
    gemm_wmma_f16<<<gemm_grid(NPATCH, HID), 256, 0, stream>>>(
        T1, HID, Wdl2[i], HID, b2, Hc, Hn, nullptr, HID, HID, HID, 0);
    half_t* t = Hc; Hc = Hn; Hn = t;
  }

  // out_patches: N=48 (padded weights to 64, ldc=64)
  gemm_wmma_f16<<<gemm_grid(NPATCH, VMODE), 256, 0, stream>>>(
      Hc, HID, Wout, HID, outb, nullptr, nullptr, Op, 64, VMODE, HID, 0);

  // ---- unpatchify + smoothing convs + residual ----
  unpatch_k<<<(2 * 3 * 65536 + 255) / 256, 256, 0, stream>>>(Op, Img);
  conv1_k<<<(2 * 8 * 65536 + 255) / 256, 256, 0, stream>>>(Img, c1w, c1b, Y1);
  conv2_k<<<(2 * 3 * 65536 + 255) / 256, 256, 0, stream>>>(Y1, c2w, c2b, Img, (float*)d_out);
}
